// Asymm_3d_spconv_segmentation_16750372455151
// MI455X (gfx1250) — matrix operand order repaired and checked by reading, not compiled, not run on hardware
//
#include <hip/hip_runtime.h>
#include <hip/hip_bf16.h>

// CDNA5 / gfx1250 sparse-conv backbone using V_WMMA_F32_16X16X32_BF16.
// Rationale (compile-only, reasoned from MI455X specs):
//  - ~7 GFLOP total, feats table (25.6MB bf16) + intermediates fit in 192MB L2
//    -> gather-bandwidth bound; bf16 activations halve gather bytes.
//  - bf16 WMMA has K=32 per instruction vs K=4 for fp32 WMMA -> matrix path
//    only pays off in bf16; fp32 accumulation + per-layer BatchNorm keeps
//    numerics sane.
//  - Weights are block-uniform: stage each tap's packed B-fragments in LDS
//    (16KB max) so the VMEM pipe is dedicated to the per-lane A gathers and
//    B operands come from ds_load_b128 in parallel.

typedef __attribute__((ext_vector_type(16))) __bf16 v16bf;
typedef __attribute__((ext_vector_type(8)))  float  v8f;

__device__ __forceinline__ unsigned short f2bf(float f) {
    unsigned int u = __float_as_uint(f);
    unsigned int r = u + 0x7FFFu + ((u >> 16) & 1u);   // round-to-nearest-even
    return (unsigned short)(r >> 16);
}

// ---------------------------------------------------------------------------
// feats fp32 [N,256] -> bf16 [N+1,256], pad row N = 0 (sentinel zero row)
// ---------------------------------------------------------------------------
__global__ void convert_feats_kernel(const float* __restrict__ x,
                                     unsigned short* __restrict__ y,
                                     int Nrows, int C) {
    int i = blockIdx.x * 256 + threadIdx.x;
    int total = (Nrows + 1) * C;
    if (i >= total) return;
    int r = i / C;
    y[i] = (r < Nrows) ? f2bf(x[i]) : (unsigned short)0;
}

// ---------------------------------------------------------------------------
// Pack W [T,Cin,Cout] fp32 into this kernel's bf16 B-fragment layout (contiguous k):
// P[((t*KC+kc)*CT+ct)*32 + lane][g]  (8 dwords per lane, contiguous)
//   lane 0-15 : col = ct*16+lane,  dword g packs K = kc*32 + 2g, 2g+1
//   lane 16-31: col = ct*16+l-16,  dword g packs K = kc*32 + 16 + 2g, +1
// ---------------------------------------------------------------------------
__global__ void pack_weights_kernel(const float* __restrict__ W,
                                    unsigned int* __restrict__ P,
                                    int T, int Cin, int Cout) {
    int KC = Cin / 32, CT = Cout / 16;
    int total = T * KC * CT * 32 * 8;
    int i = blockIdx.x * 256 + threadIdx.x;
    if (i >= total) return;
    int g    = i & 7;
    int lane = (i >> 3) & 31;
    int rest = i >> 8;                 // t*KC*CT + kc*CT + ct
    int ct = rest % CT; rest /= CT;
    int kc = rest % KC;
    int t  = rest / KC;
    int group = lane >> 4;
    int col   = ct * 16 + (lane & 15);
    int k0    = kc * 32 + group * 16 + 2 * g;
    unsigned int lo = f2bf(W[((size_t)t * Cin + k0)     * Cout + col]);
    unsigned int hi = f2bf(W[((size_t)t * Cin + k0 + 1) * Cout + col]);
    P[i] = lo | (hi << 16);
}

// ---------------------------------------------------------------------------
// Gather-GEMM sparse conv: out[m,:] = sum_t A[nbr[m,t],:] @ W[t]
// One wave = one 16-row tile x all Cout column tiles (CT accumulators).
// A fragment (contiguous k order, same as packed B): per lane two 16B global loads, K-runs
// [group*16,+8) and [group*16+8,+8) of the gathered bf16 row.
// B fragments for the current tap staged in LDS by the whole block.
// No early returns: per-tap __syncthreads() requires all 8 waves, and WMMA
// requires EXEC all-ones (tail waves compute on the sentinel row, skip store).
// ---------------------------------------------------------------------------
template <int CIN, int COUT, int TAPS>
__global__ __launch_bounds__(256) void spconv_wmma_kernel(
    const unsigned short* __restrict__ A,   // [Nsent+1, CIN] bf16, row Nsent = 0
    const int* __restrict__ nbr,            // [M, TAPS]
    const unsigned int* __restrict__ P,     // packed bf16 weights
    float* __restrict__ out,                // [M, COUT] fp32
    int M, int Nsent) {
    constexpr int KC = CIN / 32;
    constexpr int CT = COUT / 16;
    constexpr int FRAG_DW = 32 * 8;               // dwords per 32x16 B fragment
    constexpr int TAP_DW  = KC * CT * FRAG_DW;    // dwords per tap (<= 4096)
    __shared__ unsigned int ldsW[TAP_DW];

    int wave = threadIdx.x >> 5;
    int lane = threadIdx.x & 31;
    int tile = blockIdx.x * 8 + wave;             // wave-uniform
    int numTiles = (M + 15) >> 4;
    bool tileOk = tile < numTiles;
    int rowBase = tile * 16;

    int r      = rowBase + (lane & 15);
    bool rowOk = tileOk && (r < M);
    int group  = lane >> 4;

    v8f acc[CT] = {};

    for (int t = 0; t < TAPS; ++t) {
        // ---- cooperative stage of this tap's B fragments into LDS ----
        __syncthreads();                          // previous tap's reads done
        {
            const uint4* src = (const uint4*)(P + (size_t)t * TAP_DW);
            uint4* dst = (uint4*)ldsW;
#pragma unroll
            for (int i = threadIdx.x; i < TAP_DW / 4; i += 256) dst[i] = src[i];
        }
        __syncthreads();

        int idx = rowOk ? nbr[(size_t)r * TAPS + t] : Nsent;
        const unsigned short* arow = A + (size_t)idx * CIN;
        if (t + 1 < TAPS) {   // gfx1250 global_prefetch_b8 for next gathered row
            int idxn = rowOk ? nbr[(size_t)r * TAPS + t + 1] : Nsent;
            __builtin_prefetch(A + (size_t)idxn * CIN, 0, 0);
        }
#pragma unroll
        for (int kc = 0; kc < KC; ++kc) {
            union { v16bf v; uint4 q[2]; } a;
            a.q[0] = *(const uint4*)(arow + kc * 32 + group * 16);
            a.q[1] = *(const uint4*)(arow + kc * 32 + group * 16 + 8);
#pragma unroll
            for (int ct = 0; ct < CT; ++ct) {
                union { v16bf v; uint4 q[2]; } b;
                const unsigned int* pp =
                    ldsW + ((kc * CT + ct) * 32 + lane) * 8;
                b.q[0] = *(const uint4*)pp;       // ds_load_b128
                b.q[1] = *(const uint4*)(pp + 4); // ds_load_b128
                acc[ct] = __builtin_amdgcn_wmma_f32_16x16x32_bf16(
                    false, a.v, false, b.v, (short)0, acc[ct], false, false);
            }
        }
    }

    // f32 C/D layout: VGPR g <-> row M = g + 8*group, col = lane&15
#pragma unroll
    for (int ct = 0; ct < CT; ++ct) {
        int col = ct * 16 + (lane & 15);
#pragma unroll
        for (int g = 0; g < 8; ++g) {
            int row = rowBase + g + group * 8;
            if (tileOk && row < M) out[(size_t)row * COUT + col] = acc[ct][g];
        }
    }
}

// ---------------------------------------------------------------------------
// BN stats over activated values: stats[0:C)=sum(lrelu(x)), [C:2C)=sum(sq)
// blockDim.x == C (32 or 64); grid-strided over rows; 2 atomics per thread.
// ---------------------------------------------------------------------------
__global__ void bn_stats_kernel(const float* __restrict__ x, int Nrows, int C,
                                float* __restrict__ stats) {
    int c = threadIdx.x;
    float s = 0.f, s2 = 0.f;
    for (int r = blockIdx.x; r < Nrows; r += gridDim.x) {
        float v = x[(size_t)r * C + c];
        v = v > 0.f ? v : 0.01f * v;
        s += v; s2 += v * v;
    }
    atomicAdd(&stats[c], s);
    atomicAdd(&stats[C + c], s2);
}

// lrelu -> normalize; writes optional fp32 [N,C] and/or bf16 [N+1,C] (pad=0)
__global__ void bn_apply_kernel(const float* __restrict__ x,
                                const float* __restrict__ stats, int Nrows,
                                int C, float* __restrict__ yf,
                                unsigned short* __restrict__ ybf) {
    int i = blockIdx.x * 256 + threadIdx.x;
    int total = (Nrows + 1) * C;
    if (i >= total) return;
    int r = i / C, c = i - r * C;
    float o = 0.f;
    if (r < Nrows) {
        float v = x[i];
        v = v > 0.f ? v : 0.01f * v;
        float invN = 1.f / (float)Nrows;
        float m    = stats[c] * invN;
        float var  = stats[C + c] * invN - m * m;
        o = (v - m) * rsqrtf(var + 1e-5f);
        if (yf) yf[i] = o;
    }
    if (ybf) ybf[i] = f2bf(o);
}

// o = a + b; writes optional fp32 [N,C] and/or bf16 [N+1,C] (pad=0)
__global__ void add2_kernel(const float* __restrict__ a,
                            const float* __restrict__ b, int Nrows, int C,
                            float* __restrict__ yf,
                            unsigned short* __restrict__ ybf) {
    int i = blockIdx.x * 256 + threadIdx.x;
    int total = (Nrows + 1) * C;
    if (i >= total) return;
    int r = i / C;
    float o = 0.f;
    if (r < Nrows) {
        o = a[i] + b[i];
        if (yf) yf[i] = o;
    }
    if (ybf) ybf[i] = f2bf(o);
}

// ---------------------------------------------------------------------------
extern "C" void kernel_launch(void* const* d_in, const int* in_sizes, int n_in,
                              void* d_out, int out_size, void* d_ws,
                              size_t ws_size, hipStream_t stream) {
    const float* feats  = (const float*)d_in[0];
    const float* W_c1   = (const float*)d_in[1];
    const float* W_c12  = (const float*)d_in[2];
    const float* W_c2   = (const float*)d_in[3];
    const float* W_c3   = (const float*)d_in[4];
    const float* W_r1   = (const float*)d_in[5];
    const float* W_r12  = (const float*)d_in[6];
    const float* W_r2   = (const float*)d_in[7];
    const float* W_r3   = (const float*)d_in[8];
    const float* W_pool = (const float*)d_in[9];
    const int* nbr_133  = (const int*)d_in[10];
    const int* nbr_313  = (const int*)d_in[11];
    const int* nbr_pool = (const int*)d_in[12];

    const int N  = in_sizes[0] / 256;   // 50000
    const int Mp = in_sizes[12] / 27;   // pool output rows

    float* resB = (float*)d_out;                    // [Mp, 64]
    float* resA = (float*)d_out + (size_t)Mp * 64;  // [N, 64]

    // ---- workspace layout ----
    char* ws = (char*)d_ws;
    size_t off = 0;
    auto alloc = [&](size_t bytes) -> char* {
        char* p = ws + off;
        off = (off + bytes + 255) & ~(size_t)255;
        return p;
    };
    unsigned short* feats_bf = (unsigned short*)alloc((size_t)(N + 1) * 256 * 2);
    unsigned int* Pc1   = (unsigned int*)alloc((size_t)9 * 256 * 32 * 2);
    unsigned int* Pc12  = (unsigned int*)alloc((size_t)9 * 32 * 32 * 2);
    unsigned int* Pc2   = (unsigned int*)alloc((size_t)9 * 256 * 32 * 2);
    unsigned int* Pc3   = (unsigned int*)alloc((size_t)9 * 32 * 32 * 2);
    unsigned int* Pr1   = (unsigned int*)alloc((size_t)9 * 32 * 64 * 2);
    unsigned int* Pr12  = (unsigned int*)alloc((size_t)9 * 64 * 64 * 2);
    unsigned int* Pr2   = (unsigned int*)alloc((size_t)9 * 32 * 64 * 2);
    unsigned int* Pr3   = (unsigned int*)alloc((size_t)9 * 64 * 64 * 2);
    unsigned int* Ppool = (unsigned int*)alloc((size_t)27 * 64 * 64 * 2);
    float* stats   = (float*)alloc(8 * 128 * sizeof(float));  // 8 layers x (2*64)
    float* convout = (float*)alloc((size_t)N * 64 * sizeof(float));
    float* s_f32a  = (float*)alloc((size_t)N * 32 * sizeof(float));
    float* r_f32a  = (float*)alloc((size_t)N * 32 * sizeof(float));
    float* s_f32b  = (float*)alloc((size_t)N * 64 * sizeof(float));
    float* r_f32b  = (float*)alloc((size_t)N * 64 * sizeof(float));
    unsigned short* s1_bf   = (unsigned short*)alloc((size_t)(N + 1) * 32 * 2);
    unsigned short* r1_bf   = (unsigned short*)alloc((size_t)(N + 1) * 32 * 2);
    unsigned short* x1_bf   = (unsigned short*)alloc((size_t)(N + 1) * 32 * 2);
    unsigned short* s5_bf   = (unsigned short*)alloc((size_t)(N + 1) * 64 * 2);
    unsigned short* r7_bf   = (unsigned short*)alloc((size_t)(N + 1) * 64 * 2);
    unsigned short* resA_bf = (unsigned short*)alloc((size_t)(N + 1) * 64 * 2);
    (void)ws_size; (void)n_in;

    auto cdiv = [](int a, int b) { return (a + b - 1) / b; };

    // stats must start at zero every call (graph replays this memset too)
    hipMemsetAsync(stats, 0, 8 * 128 * sizeof(float), stream);

    // feats -> bf16 (+ zero sentinel row)
    convert_feats_kernel<<<cdiv((N + 1) * 256, 256), 256, 0, stream>>>(
        feats, feats_bf, N, 256);

    // pack all weights into B-fragment layout
    auto pack = [&](const float* W, unsigned int* P, int T, int Cin, int Cout) {
        int total = T * (Cin / 32) * (Cout / 16) * 256;
        pack_weights_kernel<<<cdiv(total, 256), 256, 0, stream>>>(W, P, T, Cin, Cout);
    };
    pack(W_c1, Pc1, 9, 256, 32);
    pack(W_c12, Pc12, 9, 32, 32);
    pack(W_c2, Pc2, 9, 256, 32);
    pack(W_c3, Pc3, 9, 32, 32);
    pack(W_r1, Pr1, 9, 32, 64);
    pack(W_r12, Pr12, 9, 64, 64);
    pack(W_r2, Pr2, 9, 32, 64);
    pack(W_r3, Pr3, 9, 64, 64);
    pack(W_pool, Ppool, 27, 64, 64);

    const int blksN  = cdiv(cdiv(N, 16), 8);
    const int blksMp = cdiv(cdiv(Mp, 16), 8);

    // ---- downCntx: ResContextBlock(256 -> 32) ----
    // s = act_bn(conv(feats, nbr_133, W_c1))
    spconv_wmma_kernel<256, 32, 9><<<blksN, 256, 0, stream>>>(
        feats_bf, nbr_133, Pc1, convout, N, N);
    bn_stats_kernel<<<512, 32, 0, stream>>>(convout, N, 32, stats + 0 * 128);
    bn_apply_kernel<<<cdiv((N + 1) * 32, 256), 256, 0, stream>>>(
        convout, stats + 0 * 128, N, 32, nullptr, s1_bf);
    // s = act_bn(conv(s, nbr_313, W_c12))
    spconv_wmma_kernel<32, 32, 9><<<blksN, 256, 0, stream>>>(
        s1_bf, nbr_313, Pc12, convout, N, N);
    bn_stats_kernel<<<512, 32, 0, stream>>>(convout, N, 32, stats + 1 * 128);
    bn_apply_kernel<<<cdiv((N + 1) * 32, 256), 256, 0, stream>>>(
        convout, stats + 1 * 128, N, 32, s_f32a, nullptr);
    // r = act_bn(conv(feats, nbr_313, W_c2))
    spconv_wmma_kernel<256, 32, 9><<<blksN, 256, 0, stream>>>(
        feats_bf, nbr_313, Pc2, convout, N, N);
    bn_stats_kernel<<<512, 32, 0, stream>>>(convout, N, 32, stats + 2 * 128);
    bn_apply_kernel<<<cdiv((N + 1) * 32, 256), 256, 0, stream>>>(
        convout, stats + 2 * 128, N, 32, nullptr, r1_bf);
    // r = act_bn(conv(r, nbr_133, W_c3))
    spconv_wmma_kernel<32, 32, 9><<<blksN, 256, 0, stream>>>(
        r1_bf, nbr_133, Pc3, convout, N, N);
    bn_stats_kernel<<<512, 32, 0, stream>>>(convout, N, 32, stats + 3 * 128);
    bn_apply_kernel<<<cdiv((N + 1) * 32, 256), 256, 0, stream>>>(
        convout, stats + 3 * 128, N, 32, r_f32a, nullptr);
    // x1 = r + s  (bf16 only; feeds two convs)
    add2_kernel<<<cdiv((N + 1) * 32, 256), 256, 0, stream>>>(
        r_f32a, s_f32a, N, 32, nullptr, x1_bf);

    // ---- resBlock2: ResBlock(32 -> 64) ----
    // s = act_bn(conv(x1, nbr_313, W_r1))
    spconv_wmma_kernel<32, 64, 9><<<blksN, 256, 0, stream>>>(
        x1_bf, nbr_313, Pr1, convout, N, N);
    bn_stats_kernel<<<512, 64, 0, stream>>>(convout, N, 64, stats + 4 * 128);
    bn_apply_kernel<<<cdiv((N + 1) * 64, 256), 256, 0, stream>>>(
        convout, stats + 4 * 128, N, 64, nullptr, s5_bf);
    // s = act_bn(conv(s, nbr_133, W_r12))
    spconv_wmma_kernel<64, 64, 9><<<blksN, 256, 0, stream>>>(
        s5_bf, nbr_133, Pr12, convout, N, N);
    bn_stats_kernel<<<512, 64, 0, stream>>>(convout, N, 64, stats + 5 * 128);
    bn_apply_kernel<<<cdiv((N + 1) * 64, 256), 256, 0, stream>>>(
        convout, stats + 5 * 128, N, 64, s_f32b, nullptr);
    // r = act_bn(conv(x1, nbr_133, W_r2))
    spconv_wmma_kernel<32, 64, 9><<<blksN, 256, 0, stream>>>(
        x1_bf, nbr_133, Pr2, convout, N, N);
    bn_stats_kernel<<<512, 64, 0, stream>>>(convout, N, 64, stats + 6 * 128);
    bn_apply_kernel<<<cdiv((N + 1) * 64, 256), 256, 0, stream>>>(
        convout, stats + 6 * 128, N, 64, nullptr, r7_bf);
    // r = act_bn(conv(r, nbr_313, W_r3))
    spconv_wmma_kernel<64, 64, 9><<<blksN, 256, 0, stream>>>(
        r7_bf, nbr_313, Pr3, convout, N, N);
    bn_stats_kernel<<<512, 64, 0, stream>>>(convout, N, 64, stats + 7 * 128);
    bn_apply_kernel<<<cdiv((N + 1) * 64, 256), 256, 0, stream>>>(
        convout, stats + 7 * 128, N, 64, r_f32b, nullptr);
    // resA = r + s  (fp32 -> d_out, bf16 for pool conv)
    add2_kernel<<<cdiv((N + 1) * 64, 256), 256, 0, stream>>>(
        r_f32b, s_f32b, N, 64, resA, resA_bf);

    // resB = conv(resA, nbr_pool, W_pool)   (27 taps, no act/bn)
    spconv_wmma_kernel<64, 64, 27><<<blksMp, 256, 0, stream>>>(
        resA_bf, nbr_pool, Ppool, resB, Mp, N);
}
